// FenrirNet_65034394796799
// MI455X (gfx1250) — matrix operand order repaired and checked by reading, not compiled, not run on hardware
//
#include <hip/hip_runtime.h>
#include <hip/hip_bf16.h>

// ---------------------------------------------------------------------------
// FenrirNet spiking CNN forward on gfx1250 (CDNA5).
//  conv2/conv3 -> implicit GEMM via V_WMMA_F32_16X16X32_F16 (wave32 fragments)
//  conv1       -> scalar VALU (Cin=1), fused IF + 2x2 spike-OR pool
//  FC          -> per-image dot products + IF membrane with quant clamp
// Activations/weights f16 NHWC; membranes f32 NHWC; everything L2-resident.
// ---------------------------------------------------------------------------

typedef __attribute__((ext_vector_type(16))) _Float16 v16h;
typedef __attribute__((ext_vector_type(8)))  _Float16 v8h;
typedef __attribute__((ext_vector_type(8)))  float    v8f;

#define BB   64   // batch
#define TT   30   // timesteps
#define NC   11   // classes

// ---------------------------------------------------------------------------
// fc1 quantization: scale = max|w| / (2^(4-1)-1)
// ---------------------------------------------------------------------------
__global__ __launch_bounds__(256) void absmax_k(const float* __restrict__ w,
                                                float* __restrict__ scale, int n) {
  __shared__ float red[256];
  float m = 0.0f;
  for (int i = threadIdx.x; i < n; i += 256) m = fmaxf(m, fabsf(w[i]));
  red[threadIdx.x] = m;
  __syncthreads();
  for (int s = 128; s > 0; s >>= 1) {
    if (threadIdx.x < s) red[threadIdx.x] = fmaxf(red[threadIdx.x], red[threadIdx.x + s]);
    __syncthreads();
  }
  if (threadIdx.x == 0) scale[0] = red[0] / 7.0f;
}

__global__ __launch_bounds__(256) void quant_k(const float* __restrict__ w,
                                               const float* __restrict__ scale,
                                               float* __restrict__ qw, int n) {
  int i = blockIdx.x * 256 + threadIdx.x;
  if (i >= n) return;
  const float s = scale[0];
  float q = rintf(w[i] / s);                 // round-half-even, matches jnp.round
  q = fminf(fmaxf(q, -7.0f), 7.0f);
  qw[i] = q * s;
}

// ---------------------------------------------------------------------------
// Pack conv weights (Cout,Cin,3,3) f32 -> WMMA-B layout f16:
//   out[((tap*(Cin/32)+icc)*Cout + n)*32 + icl] = w[n][icc*32+icl][tap]
// ---------------------------------------------------------------------------
__global__ __launch_bounds__(256) void pack_w_k(const float* __restrict__ w,
                                                _Float16* __restrict__ out,
                                                int Cout, int Cin) {
  int i = blockIdx.x * 256 + threadIdx.x;
  const int total = Cout * Cin * 9;
  if (i >= total) return;
  const int icl = i & 31;
  int r = i >> 5;
  const int n = r % Cout; r /= Cout;
  const int nicc = Cin >> 5;
  const int icc = r % nicc;
  const int tap = r / nicc;
  const int ic = icc * 32 + icl;
  out[i] = (_Float16)w[(n * Cin + ic) * 9 + tap];
}

// ---------------------------------------------------------------------------
// conv1 (1->32, 3x3, pad 1) + IF + 2x2 spike pool, fused.
// grid = B*128, block = 256 (32 oc x 8 pooled cells); thread owns 2x2 pixels.
// m1: NHWC f32 (B,64,64,32); s1: NHWC f16 (B,32,32,32)
// ---------------------------------------------------------------------------
__global__ __launch_bounds__(256) void conv1_if_pool_k(
    const float* __restrict__ x, const float* __restrict__ w1,
    const float* __restrict__ thr, float* __restrict__ m1,
    _Float16* __restrict__ s1, int t) {
  const int tid = threadIdx.x;
  const int oc = tid & 31;
  const int b = blockIdx.x >> 7;
  const int cell = (blockIdx.x & 127) * 8 + (tid >> 5);
  const int yp = cell >> 5, xp = cell & 31;

  float w[9];
#pragma unroll
  for (int k = 0; k < 9; ++k) w[k] = w1[oc * 9 + k];
  const float th = thr[oc];
  const float* xin = x + ((size_t)b * TT + t) * 4096;

  float pooled = 0.0f;
#pragma unroll
  for (int i = 0; i < 2; ++i) {
#pragma unroll
    for (int j = 0; j < 2; ++j) {
      const int y = 2 * yp + i, x0 = 2 * xp + j;
      float cur = 0.0f;
#pragma unroll
      for (int ky = 0; ky < 3; ++ky) {
        const int yy = y + ky - 1;
        if ((unsigned)yy < 64u) {
#pragma unroll
          for (int kx = 0; kx < 3; ++kx) {
            const int xx = x0 + kx - 1;
            if ((unsigned)xx < 64u) cur = fmaf(xin[yy * 64 + xx], w[ky * 3 + kx], cur);
          }
        }
      }
      const size_t midx = ((size_t)(b * 64 + y) * 64 + x0) * 32 + oc;
      float mv = m1[midx] + cur;
      const float s = (mv > th) ? 1.0f : 0.0f;
      m1[midx] = mv * (1.0f - s);
      pooled = fmaxf(pooled, s);
    }
  }
  s1[((size_t)(b * 32 + yp) * 32 + xp) * 32 + oc] = (_Float16)pooled;
}

// ---------------------------------------------------------------------------
// Implicit-GEMM conv (3x3, pad 1, stride 1) + integrate-and-fire, WMMA f16.
//   M = HS*WS spatial, N = COUT, K = 9*CIN in chunks of 32 (one tap x 32 ic)
// 8 waves/block, one 16x16 output tile per wave; full K in registers -> IF.
// sin NHWC f16 (B,HS,WS,CIN); wh packed B-layout; mem/sfull NHWC (B,HS,WS,COUT)
// ---------------------------------------------------------------------------
template <int CIN, int COUT, int HS, int WS>
__global__ __launch_bounds__(256) void conv_if_wmma_k(
    const _Float16* __restrict__ sin, const _Float16* __restrict__ wh,
    const float* __restrict__ thr, float* __restrict__ mem,
    _Float16* __restrict__ sfull) {
  constexpr int MT = (HS * WS) / 16;
  constexpr int NT = COUT / 16;
  constexpr int ICC = CIN / 32;

  const int lane = threadIdx.x & 31;
  const int wid = blockIdx.x * 8 + (threadIdx.x >> 5);
  const int b = wid / (MT * NT);
  const int rr0 = wid % (MT * NT);
  const int mt = rr0 / NT;
  const int nt = rr0 % NT;

  const int mrow = mt * 16 + (lane & 15);  // A-matrix row owned by this lane
  const int y0 = mrow / WS, x0 = mrow % WS;
  const int kq = lane >> 4;                // which K half this lane holds
  const int n = nt * 16 + (lane & 15);     // B/D column owned by this lane

  __builtin_prefetch(wh + (size_t)nt * 16 * 32, 0, 1);  // warm packed weights

  v8f acc;
#pragma unroll
  for (int i = 0; i < 8; ++i) acc[i] = 0.0f;

#pragma unroll
  for (int tap = 0; tap < 9; ++tap) {
    const int dy = tap / 3 - 1, dx = tap % 3 - 1;
    const int y = y0 + dy, x = x0 + dx;
    const bool oob = ((unsigned)y >= (unsigned)HS) || ((unsigned)x >= (unsigned)WS);
#pragma unroll
    for (int icc = 0; icc < ICC; ++icc) {
      // A fragment: lane<16 -> K 0..7 & 16..23 ; lane>=16 -> K 8..15 & 24..31
      v16h a;
      if (!oob) {
        const _Float16* pa =
            sin + (((size_t)(b * HS + y) * WS + x) * CIN + icc * 32 + kq * 8);
        const v8h lo = *(const v8h*)pa;
        const v8h hi = *(const v8h*)(pa + 16);
#pragma unroll
        for (int i = 0; i < 8; ++i) { a[i] = lo[i]; a[i + 8] = hi[i]; }
      } else {
#pragma unroll
        for (int i = 0; i < 16; ++i) a[i] = (_Float16)0.0f;
      }
      // B fragment: lane<16 -> K 0..7 & 16..23 ; lane>=16 -> K 8..15 & 24..31
      const int chunk = tap * ICC + icc;
      const _Float16* pb = wh + (((size_t)chunk * COUT + n) * 32 + kq * 8);
      const v8h blo = *(const v8h*)pb;
      const v8h bhi = *(const v8h*)(pb + 16);
      v16h bf;
#pragma unroll
      for (int i = 0; i < 8; ++i) { bf[i] = blo[i]; bf[i + 8] = bhi[i]; }

      acc = __builtin_amdgcn_wmma_f32_16x16x32_f16(
          false, a, false, bf, (short)0, acc, false, false);
    }
  }

  // Integrate-and-fire with zero reset; D layout: VGPR r -> M=r (lanes<16) / 8+r
  const float th = thr[n];
#pragma unroll
  for (int r = 0; r < 8; ++r) {
    const int ml = (lane < 16) ? r : r + 8;
    const int m = mt * 16 + ml;
    const int yy = m / WS, xx = m % WS;
    const size_t idx = ((size_t)(b * HS + yy) * WS + xx) * COUT + n;
    float mv = mem[idx] + acc[r];
    const float s = (mv > th) ? 1.0f : 0.0f;
    mem[idx] = mv * (1.0f - s);
    sfull[idx] = (_Float16)s;
  }
}

// ---------------------------------------------------------------------------
// 2x2 spike max-pool (OR), NHWC f16 -> NHWC f16
// ---------------------------------------------------------------------------
__global__ __launch_bounds__(256) void pool2x2_k(const _Float16* __restrict__ in,
                                                 _Float16* __restrict__ out,
                                                 int C, int Hp, int Wp, int total) {
  int i = blockIdx.x * 256 + threadIdx.x;
  if (i >= total) return;
  const int c = i % C;
  int r = i / C;
  const int xp = r % Wp; r /= Wp;
  const int yp = r % Hp;
  const int b = r / Hp;
  const int Ws = Wp * 2;
  const size_t base = ((size_t)(b * Hp * 2 + yp * 2) * Ws + xp * 2) * C + c;
  float v = (float)in[base];
  v = fmaxf(v, (float)in[base + C]);
  v = fmaxf(v, (float)in[base + (size_t)Ws * C]);
  v = fmaxf(v, (float)in[base + (size_t)Ws * C + C]);
  out[i] = (_Float16)v;
}

// ---------------------------------------------------------------------------
// Quantized FC + output IF neuron (clamp / add / spike / reset / beta).
// s3 NHWC (B,8,8,128) f16; qw (11,8192) in NCHW-flatten order.
// One block per image.
// ---------------------------------------------------------------------------
__global__ __launch_bounds__(256) void fc_if_k(
    const _Float16* __restrict__ s3, const float* __restrict__ qw,
    const float* __restrict__ scale, const float* __restrict__ betap,
    const float* __restrict__ thrp, float* __restrict__ fm,
    float* __restrict__ out, int t) {
  __shared__ float red[256];
  const int b = blockIdx.x;
  const int tid = threadIdx.x;

  float part[NC];
#pragma unroll
  for (int n = 0; n < NC; ++n) part[n] = 0.0f;

  const _Float16* sp = s3 + (size_t)b * 8192;
  for (int i = tid; i < 8192; i += 256) {
    const float v = (float)sp[i];
    if (v != 0.0f) {  // spikes are sparse binary
      // NHWC flat i = (y*8+x)*128 + c  ->  NCHW flat j = c*64 + (y*8+x)
      const int j = (i & 127) * 64 + (i >> 7);
#pragma unroll
      for (int n = 0; n < NC; ++n) part[n] = fmaf(v, qw[n * 8192 + j], part[n]);
    }
  }

  float sums[NC];
  for (int n = 0; n < NC; ++n) {
    red[tid] = part[n];
    __syncthreads();
    for (int s = 128; s > 0; s >>= 1) {
      if (tid < s) red[tid] += red[tid + s];
      __syncthreads();
    }
    if (tid == 0) sums[n] = red[0];
    __syncthreads();
  }

  if (tid == 0) {
    const float lim = 64.0f * scale[0];
    const float beta = fminf(fmaxf(betap[0], 0.0f), 1.0f);
    const float th = thrp[0];
#pragma unroll
    for (int n = 0; n < NC; ++n) {
      float f = fm[b * NC + n];
      f = fminf(fmaxf(f, -lim), lim);  // mem_clamp
      f = f + sums[n];
      const float s4 = ((f - th) > 0.0f) ? 1.0f : 0.0f;
      f = f * (1.0f - s4);             // zero reset
      f = f * beta;                    // beta_clamp
      fm[b * NC + n] = f;
      out[((size_t)t * BB + b) * NC + n] = s4;
    }
  }
}

// ---------------------------------------------------------------------------
extern "C" void kernel_launch(void* const* d_in, const int* in_sizes, int n_in,
                              void* d_out, int out_size, void* d_ws, size_t ws_size,
                              hipStream_t stream) {
  (void)in_sizes; (void)n_in; (void)out_size; (void)ws_size;

  const float* x    = (const float*)d_in[0];  // (64,30,1,64,64)
  const float* w1   = (const float*)d_in[1];  // (32,1,3,3)
  const float* w2   = (const float*)d_in[2];  // (64,32,3,3)
  const float* w3   = (const float*)d_in[3];  // (128,64,3,3)
  const float* fcw  = (const float*)d_in[4];  // (11,8192)
  const float* beta = (const float*)d_in[5];
  const float* fthr = (const float*)d_in[6];
  const float* thr1 = (const float*)d_in[7];
  const float* thr2 = (const float*)d_in[8];
  const float* thr3 = (const float*)d_in[9];
  float* out = (float*)d_out;                 // (30,64,11)

  // Workspace carve-up (256B aligned blocks; membranes contiguous for memset)
  char* p = (char*)d_ws;
  auto carve = [&](size_t bytes) -> char* {
    char* r = p;
    p += (bytes + 255) & ~(size_t)255;
    return r;
  };
  float*     scale  = (float*)carve(4);
  float*     qw     = (float*)carve((size_t)NC * 8192 * 4);
  _Float16*  w2h    = (_Float16*)carve((size_t)9 * 64 * 32 * 2);
  _Float16*  w3h    = (_Float16*)carve((size_t)18 * 128 * 32 * 2);
  float*     m1     = (float*)carve((size_t)BB * 64 * 64 * 32 * 4);   // NHWC
  float*     m2     = (float*)carve((size_t)BB * 32 * 32 * 64 * 4);
  float*     m3     = (float*)carve((size_t)BB * 16 * 16 * 128 * 4);
  float*     fm     = (float*)carve((size_t)BB * NC * 4);
  _Float16*  s1     = (_Float16*)carve((size_t)BB * 32 * 32 * 32 * 2);
  _Float16*  s2full = (_Float16*)carve((size_t)BB * 32 * 32 * 64 * 2);
  _Float16*  s2     = (_Float16*)carve((size_t)BB * 16 * 16 * 64 * 2);
  _Float16*  s3full = (_Float16*)carve((size_t)BB * 16 * 16 * 128 * 2);
  _Float16*  s3     = (_Float16*)carve((size_t)BB * 8 * 8 * 128 * 2);

  // Zero all membrane state (m1..fm are contiguous 256B-multiple blocks)
  const size_t memb_bytes =
      ((size_t)BB * 64 * 64 * 32 + (size_t)BB * 32 * 32 * 64 +
       (size_t)BB * 16 * 16 * 128) * 4 + (((size_t)BB * NC * 4 + 255) & ~(size_t)255);
  hipMemsetAsync(m1, 0, memb_bytes, stream);

  // Quantize fc weights; pack conv weights into WMMA-B f16 layout
  absmax_k<<<1, 256, 0, stream>>>(fcw, scale, NC * 8192);
  quant_k<<<(NC * 8192 + 255) / 256, 256, 0, stream>>>(fcw, scale, qw, NC * 8192);
  pack_w_k<<<(64 * 32 * 9 + 255) / 256, 256, 0, stream>>>(w2, w2h, 64, 32);
  pack_w_k<<<(128 * 64 * 9 + 255) / 256, 256, 0, stream>>>(w3, w3h, 128, 64);

  for (int t = 0; t < TT; ++t) {
    conv1_if_pool_k<<<BB * 128, 256, 0, stream>>>(x, w1, thr1, m1, s1, t);

    // conv2: M=1024 (64 tiles), N=64 (4 tiles) -> 16384 waves / 8 = 2048 blocks
    conv_if_wmma_k<32, 64, 32, 32><<<2048, 256, 0, stream>>>(s1, w2h, thr2, m2, s2full);
    pool2x2_k<<<(BB * 16 * 16 * 64 + 255) / 256, 256, 0, stream>>>(
        s2full, s2, 64, 16, 16, BB * 16 * 16 * 64);

    // conv3: M=256 (16 tiles), N=128 (8 tiles) -> 8192 waves / 8 = 1024 blocks
    conv_if_wmma_k<64, 128, 16, 16><<<1024, 256, 0, stream>>>(s2, w3h, thr3, m3, s3full);
    pool2x2_k<<<(BB * 8 * 8 * 128 + 255) / 256, 256, 0, stream>>>(
        s3full, s3, 128, 8, 8, BB * 8 * 8 * 128);

    fc_if_k<<<BB, 256, 0, stream>>>(s3, qw, scale, beta, fthr, fm, out, t);
  }
}
